// TwoStageNet_53111565582355
// MI455X (gfx1250) — matrix operand order repaired and checked by reading, not compiled, not run on hardware
//
#include <hip/hip_runtime.h>
#include <hip/hip_bf16.h>

// ---------------- problem constants ----------------
#define BS      8
#define CC      256
#define HH      128
#define WW      128
#define HW      (HH * WW)          // 16384
#define ANCH    9
#define NHM     (BS * ANCH * HW)   // 1179648
#define KTOP    100
#define NDET    (BS * KTOP)        // 800
#define FEATR   (CC * ANCH)        // 2304 (fc1 input row length)
#define HID     1024
#define KSTEPS  (FEATR / 32)       // 72

// ---------------- output layout (floats) ----------------
#define O_HM     0
#define O_WH1    1179648
#define O_OFF1   1182848
#define O_S1CLS  1184448
#define O_WH2    1185248
#define O_OFF2   1188448
#define O_S2CLS  1190048
#define O_S2WH   1198048
#define O_INDS   1201248
#define O_VAL    1202048

typedef __attribute__((ext_vector_type(16))) __bf16 v16bf;
typedef __attribute__((ext_vector_type(8)))  __bf16 v8bf;
typedef __attribute__((ext_vector_type(8)))  float  v8f;
typedef __attribute__((ext_vector_type(4)))  int    v4i;

#ifndef __has_builtin
#define __has_builtin(x) 0
#endif
#if __has_builtin(__builtin_amdgcn_global_load_async_to_lds_b128)
#define USE_ASYNC_LDS 1
#else
#define USE_ASYNC_LDS 0
#endif

// f32 -> bf16 (round-to-nearest-even), storage as u16 (host-safe pointers)
__device__ __forceinline__ unsigned short f2bfu(float f) {
    unsigned u = __builtin_bit_cast(unsigned, f);
    u += 0x7FFFu + ((u >> 16) & 1u);
    return (unsigned short)(u >> 16);
}

// ---------------- kernel 1: hm -> out and hm -> ws (mutable copy for topk) ----
__global__ void copy_hm_kernel(const float4* __restrict__ hm,
                               float4* __restrict__ out_hm,
                               float4* __restrict__ ws_hm) {
    int i = blockIdx.x * blockDim.x + threadIdx.x;   // NHM/4 = 294912 total
    float4 v = hm[i];
    out_hm[i] = v;
    ws_hm[i]  = v;
}

// ---------------- kernel 2: per-batch iterative top-k (argmax x100) ----------
__global__ void __launch_bounds__(1024) topk_kernel(float* __restrict__ data,
                                                    float* __restrict__ out_val,
                                                    int*   __restrict__ out_idx) {
    const int N = ANCH * HW;                    // 147456 per batch
    const int b = blockIdx.x;
    float* d = data + (size_t)b * N;
    __shared__ float sv[1024];
    __shared__ int   si[1024];
    for (int it = 0; it < KTOP; ++it) {
        float best = -3.402823466e38f;
        int   bidx = 0;
        for (int i = threadIdx.x; i < N; i += 1024) {
            float v = d[i];
            if (v > best) { best = v; bidx = i; }
        }
        sv[threadIdx.x] = best;
        si[threadIdx.x] = bidx;
        __syncthreads();
        for (int s = 512; s > 0; s >>= 1) {
            if ((int)threadIdx.x < s) {
                if (sv[threadIdx.x + s] > sv[threadIdx.x]) {
                    sv[threadIdx.x] = sv[threadIdx.x + s];
                    si[threadIdx.x] = si[threadIdx.x + s];
                }
            }
            __syncthreads();
        }
        if (threadIdx.x == 0) {
            out_val[b * KTOP + it] = sv[0];
            out_idx[b * KTOP + it] = si[0];
            d[si[0]] = -3.402823466e38f;        // mask winner
        }
        __syncthreads();
    }
}

// ---------------- kernel 3: decode detections, gather wh/offset, make boxes --
__global__ void prep_kernel(const float* __restrict__ wh,
                            const float* __restrict__ offs,
                            const int*   __restrict__ tidx,
                            float* __restrict__ out,
                            float* __restrict__ boxes) {
    int r = blockIdx.x * blockDim.x + threadIdx.x;
    if (r >= NDET) return;
    int b   = r / KTOP;
    int ti  = tidx[r];
    int s1  = ti % ANCH;
    int pos = (ti / ANCH) % HW;
    int ind = pos + b * HW;
    // reference select_tensor semantics: row ind of (bs,hw,9,c) flat -> batch 0
    int a = ind % ANCH;
    int p = ind / ANCH;
    float w0 = wh[(a * 4 + 0) * HW + p];
    float w1 = wh[(a * 4 + 1) * HW + p];
    float w2 = wh[(a * 4 + 2) * HW + p];
    float w3 = wh[(a * 4 + 3) * HW + p];
    float o0 = offs[(a * 2 + 0) * HW + p];
    float o1 = offs[(a * 2 + 1) * HW + p];
    float xs = (float)(ind % WW) + o0;
    float ys = (float)(ind / WW) + o1;
    boxes[r * 4 + 0] = xs + w0;
    boxes[r * 4 + 1] = ys + w1;
    boxes[r * 4 + 2] = xs + w2;
    boxes[r * 4 + 3] = ys + w3;
    out[O_WH1  + r * 4 + 0] = w0;  out[O_WH2  + r * 4 + 0] = w0;
    out[O_WH1  + r * 4 + 1] = w1;  out[O_WH2  + r * 4 + 1] = w1;
    out[O_WH1  + r * 4 + 2] = w2;  out[O_WH2  + r * 4 + 2] = w2;
    out[O_WH1  + r * 4 + 3] = w3;  out[O_WH2  + r * 4 + 3] = w3;
    out[O_OFF1 + r * 2 + 0] = o0;  out[O_OFF2 + r * 2 + 0] = o0;
    out[O_OFF1 + r * 2 + 1] = o1;  out[O_OFF2 + r * 2 + 1] = o1;
    out[O_S1CLS + r] = (float)s1;
    out[O_INDS  + r] = (float)ind;
}

// ---------------- kernel 4: 3x3 ROI-align -> bf16 rows (lane = channel) -----
__global__ void __launch_bounds__(256) roi_kernel(const float* __restrict__ feat,
                                                  const float* __restrict__ boxes,
                                                  unsigned short* __restrict__ roiB) {
    int r = blockIdx.x;
    int c = threadIdx.x;
    int b = r / KTOP;
    float x1 = boxes[r * 4 + 0], y1 = boxes[r * 4 + 1];
    float x2 = boxes[r * 4 + 2], y2 = boxes[r * 4 + 3];
    float bw = fmaxf(x2 - x1, 1.0f) * (1.0f / 3.0f);
    float bh = fmaxf(y2 - y1, 1.0f) * (1.0f / 3.0f);
    const float* fb = feat + ((size_t)b * CC + c) * HW;
    #pragma unroll
    for (int g = 0; g < 9; ++g) {
        int gy = g / 3, gx = g % 3;
        float ysf = y1 + ((float)gy + 0.5f) * bh;
        float xsf = x1 + ((float)gx + 0.5f) * bw;
        bool invalid = (ysf < -1.0f) || (ysf > (float)HH) ||
                       (xsf < -1.0f) || (xsf > (float)WW);
        float y = fminf(fmaxf(ysf, 0.0f), (float)(HH - 1));
        float x = fminf(fmaxf(xsf, 0.0f), (float)(WW - 1));
        int y0 = (int)floorf(y), x0 = (int)floorf(x);
        int y1i = min(y0 + 1, HH - 1), x1i = min(x0 + 1, WW - 1);
        float ly = y - (float)y0, lx = x - (float)x0;
        float hy = 1.0f - ly,     hx = 1.0f - lx;
        float v = hy * hx * fb[y0  * WW + x0 ] + hy * lx * fb[y0  * WW + x1i]
                + ly * hx * fb[y1i * WW + x0 ] + ly * lx * fb[y1i * WW + x1i];
        roiB[(size_t)r * FEATR + c * 9 + g] = f2bfu(invalid ? 0.0f : v);
    }
}

// ---------------- kernel 5a: w_fc1 (2304x1024 f32) -> wT (1024x2304 bf16) ----
__global__ void convert_w_kernel(const float* __restrict__ w1,
                                 unsigned short* __restrict__ wT) {
    int idx = blockIdx.x * blockDim.x + threadIdx.x;  // FEATR*HID total
    int k = idx / HID, n = idx % HID;                 // coalesced read along n
    wT[(size_t)n * FEATR + k] = f2bfu(w1[idx]);
}

// ---------------- kernel 5b: FC1 (800x2304 @ 2304x1024) via bf16 WMMA -------
// block = 256 threads = 8 wave32s; each wave -> 16x32 tile (2 accumulators).
// A tile (16x32 bf16) double-buffered in LDS via async global->LDS copies.
// grid = (HID/256, NDET/16) = (4, 50)
__global__ void __launch_bounds__(256) fc1_wmma_kernel(const unsigned short* __restrict__ roiB,
                                                       const unsigned short* __restrict__ wT,
                                                       const float* __restrict__ b1,
                                                       float* __restrict__ hdn) {
    const int tid   = threadIdx.x;
    const int lane  = tid & 31;
    const int wave  = tid >> 5;
    const int half  = lane >> 4;     // half-wave selects K sub-block
    const int l16   = lane & 15;     // A row / B col / D col
    const int mBase = blockIdx.y * 16;
    const int nBase = blockIdx.x * 256 + wave * 32;

    __shared__ __align__(16) unsigned short sA[2][16 * 40];  // 40-elem row pitch

    auto stage = [&](int buf, int k0) {
        if (tid < 64) {                       // waves 0 & 1, EXEC uniform
            int row = tid >> 2, seg = tid & 3;
            const unsigned short* g =
                roiB + (size_t)(mBase + row) * FEATR + k0 + seg * 8;
            unsigned short* l = &sA[buf][row * 40 + seg * 8];
#if USE_ASYNC_LDS
            // param 0: v4i* (global src), param 1: v4i* (LDS dst) -- HIP generic
            // pointers implicitly convert to the CUDA language address spaces.
            __builtin_amdgcn_global_load_async_to_lds_b128((v4i*)g, (v4i*)l, 0, 0);
#else
            *(v8bf*)l = *(const v8bf*)g;
#endif
        }
    };

    stage(0, 0);
#if USE_ASYNC_LDS
    asm volatile("s_wait_asynccnt 0x0" ::: "memory");
#endif
    __syncthreads();

    v8f acc0 = {}, acc1 = {};
    const unsigned short* brow0 = wT + (size_t)(nBase + l16) * FEATR;
    const unsigned short* brow1 = wT + (size_t)(nBase + 16 + l16) * FEATR;

    for (int kc = 0; kc < KSTEPS; ++kc) {
        const int cur = kc & 1;
        if (kc + 1 < KSTEPS) stage(cur ^ 1, (kc + 1) * 32);
        const int k0 = kc * 32;
        __builtin_prefetch(brow0 + k0 + 256, 0, 0);   // global_prefetch_b8

        // A fragment: e -> K = k0 + half*16 + e (matches B); rows = l16
        union { v16bf v; v8bf h[2]; } A, B0, B1;
        A.h[0]  = *(const v8bf*)&sA[cur][l16 * 40 + half * 16];
        A.h[1]  = *(const v8bf*)&sA[cur][l16 * 40 + half * 16 + 8];
        // B fragment from transposed weights: e -> K = k0 + half*16 + e
        B0.h[0] = *(const v8bf*)(brow0 + k0 + half * 16);
        B0.h[1] = *(const v8bf*)(brow0 + k0 + half * 16 + 8);
        B1.h[0] = *(const v8bf*)(brow1 + k0 + half * 16);
        B1.h[1] = *(const v8bf*)(brow1 + k0 + half * 16 + 8);

        acc0 = __builtin_amdgcn_wmma_f32_16x16x32_bf16(
                   false, A.v, false, B0.v, (short)0, acc0, false, false);
        acc1 = __builtin_amdgcn_wmma_f32_16x16x32_bf16(
                   false, A.v, false, B1.v, (short)0, acc1, false, false);

#if USE_ASYNC_LDS
        asm volatile("s_wait_asynccnt 0x0" ::: "memory");
#endif
        __syncthreads();
    }

    // D layout: VGPR i -> M = half*8 + i, N = l16
    const int n0 = nBase + l16;
    const int n1 = nBase + 16 + l16;
    const float bb0 = b1[n0], bb1 = b1[n1];
    #pragma unroll
    for (int i = 0; i < 8; ++i) {
        int m = mBase + half * 8 + i;
        float v0 = acc0[i] + bb0;
        float v1 = acc1[i] + bb1;
        hdn[(size_t)m * HID + n0] = v0 > 0.0f ? v0 : 0.0f;
        hdn[(size_t)m * HID + n1] = v1 > 0.0f ? v1 : 0.0f;
    }
}

// ---------------- kernel 6: heads (800x1024 @ 1024x10 and 1024x4) -----------
__global__ void heads_kernel(const float* __restrict__ hdn,
                             const float* __restrict__ wcls, const float* __restrict__ bcls,
                             const float* __restrict__ wwh,  const float* __restrict__ bwh,
                             float* __restrict__ out) {
    int idx = blockIdx.x * blockDim.x + threadIdx.x;
    if (idx >= NDET * 14) return;
    int r = idx / 14, j = idx % 14;
    const float* hrow = hdn + (size_t)r * HID;
    if (j < 10) {
        float s = bcls[j];
        for (int kk = 0; kk < HID; ++kk) s += hrow[kk] * wcls[kk * 10 + j];
        out[O_S2CLS + r * 10 + j] = s;
    } else {
        int jj = j - 10;
        float s = bwh[jj];
        for (int kk = 0; kk < HID; ++kk) s += hrow[kk] * wwh[kk * 4 + jj];
        out[O_S2WH + r * 4 + jj] = s;
    }
}

extern "C" void kernel_launch(void* const* d_in, const int* in_sizes, int n_in,
                              void* d_out, int out_size, void* d_ws, size_t ws_size,
                              hipStream_t stream) {
    const float* feat   = (const float*)d_in[0];
    const float* hm     = (const float*)d_in[1];
    const float* wh     = (const float*)d_in[2];
    const float* offset = (const float*)d_in[3];
    const float* w_fc1  = (const float*)d_in[4];
    const float* b_fc1  = (const float*)d_in[5];
    const float* w_cls  = (const float*)d_in[6];
    const float* b_cls  = (const float*)d_in[7];
    const float* w_wh   = (const float*)d_in[8];
    const float* b_wh   = (const float*)d_in[9];
    float* out = (float*)d_out;

    // workspace carve-up
    float*          ws_hm    = (float*)d_ws;                          // NHM f32
    int*            ws_idx   = (int*)(ws_hm + NHM);                   // NDET i32
    float*          ws_boxes = (float*)(ws_idx + NDET);               // NDET*4 f32
    float*          ws_hdn   = ws_boxes + NDET * 4;                   // NDET*HID f32
    unsigned short* ws_roiB  = (unsigned short*)(ws_hdn + (size_t)NDET * HID);   // NDET*FEATR bf16
    unsigned short* ws_wT    = ws_roiB + (size_t)NDET * FEATR;        // HID*FEATR bf16 (transposed)

    copy_hm_kernel<<<NHM / 4 / 256, 256, 0, stream>>>(
        (const float4*)hm, (float4*)(out + O_HM), (float4*)ws_hm);

    topk_kernel<<<BS, 1024, 0, stream>>>(ws_hm, out + O_VAL, ws_idx);

    prep_kernel<<<(NDET + 127) / 128, 128, 0, stream>>>(wh, offset, ws_idx, out, ws_boxes);

    roi_kernel<<<NDET, 256, 0, stream>>>(feat, ws_boxes, ws_roiB);

    convert_w_kernel<<<(FEATR * HID) / 256, 256, 0, stream>>>(w_fc1, ws_wT);

    fc1_wmma_kernel<<<dim3(HID / 256, NDET / 16), 256, 0, stream>>>(
        ws_roiB, ws_wT, b_fc1, ws_hdn);

    heads_kernel<<<(NDET * 14 + 255) / 256, 256, 0, stream>>>(
        ws_hdn, w_cls, b_cls, w_wh, b_wh, out);
}
